// ASGLoss_11407433138190
// MI455X (gfx1250) — matrix operand order repaired and checked by reading, not compiled, not run on hardware
//
#include <hip/hip_runtime.h>

typedef __attribute__((ext_vector_type(16))) _Float16 v16h;
typedef __attribute__((ext_vector_type(8)))  float    v8f;

#define T_LEN 2048
#define N_CLS 64
#define L_TGT 256
#define NEGINF (-1e30f)
#define LOG2E  1.44269504088896340736f
#define LN2    0.69314718055994530942f

// raw v_exp_f32 / v_log_f32 (base-2), bypassing ocml denorm-fixup paths
#define EXP2(x) __builtin_amdgcn_exp2f(x)
#define LOG2(x) __builtin_amdgcn_logf(x)

// ---------------------------------------------------------------------------
// Prep: build E = exp(trans) directly in the WMMA A-fragment register layout.
// Element index = ((w*2 + f)*32 + lane)*16 + j, so the main kernel's lane just
// does one contiguous 32-byte load per fragment.
// 16-bit A 16x32 layout (ISA 7.12.2): lane<16 -> K = {0..7, 16..23},
// lane>=16 -> K = {8..15, 24..31}; row M = lane%16.
// ---------------------------------------------------------------------------
__global__ __launch_bounds__(256) void asg_prep_eA(const float* __restrict__ trans,
                                                   _Float16* __restrict__ eA) {
    int tid = blockIdx.x * 256 + threadIdx.x;
    if (tid >= 4096) return;
    int j = tid & 15;
    int L = (tid >> 4) & 31;
    int f = (tid >> 9) & 1;
    int w = tid >> 10;
    int row    = w * 16 + (L & 15);
    int halfg  = L >> 4;
    int localK = halfg * 8 + (j < 8 ? j : j + 8);
    int K      = f * 32 + localK;
    eA[tid] = (_Float16)__expf(trans[row * N_CLS + K]);
}

// ---------------------------------------------------------------------------
// fcc: full-lattice forward, entirely in the log2 domain.
// 8 blocks x 128 threads (4 waves); block = 16-batch column tile; wave w
// computes rows n in [16w, 16w+16).  Per step:
//   S = E x P  via two v_wmma_f32_16x16x32_f16 (K=64)
//   beta = x*log2e + log2(S) + betamax
// x_t is software-prefetched one iteration ahead (issued before the barriers)
// so HBM latency hides under the reduction + barriers + WMMA of the step.
// ---------------------------------------------------------------------------
__global__ __launch_bounds__(128) void asg_fcc(const float* __restrict__ input,
                                               const _Float16* __restrict__ eA,
                                               float* __restrict__ out) {
    __shared__ _Float16 pLDS[16][72];   // [local batch][k] (padded rows, 16B-aligned)
    __shared__ float    red[4][2][16];  // [wave][lane-half][local batch]

    const int tid  = threadIdx.x;
    const int w    = tid >> 5;
    const int lane = tid & 31;
    const int half = lane >> 4;
    const int bl   = lane & 15;              // local batch = B/D column
    const int n0   = w * 16 + half * 8;      // first of 8 output rows this lane owns

    // Persistent A fragments (rows 16w..16w+15 of E, K=0..31 and K=32..63)
    const v16h* ea = (const v16h*)eA;
    v16h a0 = ea[(w * 2 + 0) * 32 + lane];
    v16h a1 = ea[(w * 2 + 1) * 32 + lane];

    const float* xb = input + ((size_t)(blockIdx.x * 16 + bl)) * T_LEN * N_CLS;

    float beta[8];   // alpha * log2e
    {
        const float4* x4 = (const float4*)(xb + n0);
        float4 lo = x4[0], hi = x4[1];
        beta[0] = lo.x * LOG2E; beta[1] = lo.y * LOG2E;
        beta[2] = lo.z * LOG2E; beta[3] = lo.w * LOG2E;
        beta[4] = hi.x * LOG2E; beta[5] = hi.y * LOG2E;
        beta[6] = hi.z * LOG2E; beta[7] = hi.w * LOG2E;
    }

    // prime the x-pipeline with the row for t = 1
    float4 xlo, xhi;
    {
        const float4* x4 = (const float4*)(xb + (size_t)1 * N_CLS + n0);
        xlo = x4[0]; xhi = x4[1];
    }

    for (int t = 1; t < T_LEN; ++t) {
        // 0) issue prefetch of x for t+1 NOW (consumed next iteration); the
        //    barriers below keep it from sinking, so it gets a full iteration
        //    of latency hiding.  Clamped so the last iteration re-reads t.
        int tn = (t + 1 < T_LEN) ? (t + 1) : (T_LEN - 1);
        const float4* xn4 = (const float4*)(xb + (size_t)tn * N_CLS + n0);
        float4 nlo = xn4[0], nhi = xn4[1];

        // 1) per-batch max over all 64 classes (cross-wave via LDS)
        float m = beta[0];
        #pragma unroll
        for (int r = 1; r < 8; ++r) m = fmaxf(m, beta[r]);
        red[w][half][bl] = m;
        __syncthreads();
        float bmax = NEGINF;
        #pragma unroll
        for (int ww = 0; ww < 4; ++ww)
            #pragma unroll
            for (int h = 0; h < 2; ++h) bmax = fmaxf(bmax, red[ww][h][bl]);

        // 2) P[m,b] = exp2(beta - bmax) -> f16, store in B-fragment order
        union { _Float16 h[8]; float4 f4; } pk;
        #pragma unroll
        for (int r = 0; r < 8; ++r)
            pk.h[r] = (_Float16)EXP2(beta[r] - bmax);
        *(float4*)&pLDS[bl][n0] = pk.f4;
        __syncthreads();

        // 3) Load B fragments: lane = column bl, K = {half*8..+7, 16+half*8..+7} (+32 for frag1)
        v16h b0, b1;
        {
            const float4* p0 = (const float4*)&pLDS[bl][half * 8];
            ((float4*)&b0)[0] = p0[0];
            ((float4*)&b0)[1] = p0[2];
            const float4* p1 = (const float4*)&pLDS[bl][32 + half * 8];
            ((float4*)&b1)[0] = p1[0];
            ((float4*)&b1)[1] = p1[2];
        }
        v8f c = {};
        c = __builtin_amdgcn_wmma_f32_16x16x32_f16(false, a0, false, b0,
                                                   (short)0, c, false, false);
        c = __builtin_amdgcn_wmma_f32_16x16x32_f16(false, a1, false, b1,
                                                   (short)0, c, false, false);

        // 4) beta = x*log2e + log2(S) + bmax  (S in [~0.5, ~100]: raw log2 safe)
        float xv[8] = {xlo.x, xlo.y, xlo.z, xlo.w, xhi.x, xhi.y, xhi.z, xhi.w};
        #pragma unroll
        for (int r = 0; r < 8; ++r)
            beta[r] = fmaf(xv[r], LOG2E, LOG2(c[r]) + bmax);

        xlo = nlo; xhi = nhi;
    }

    // Final: out[b] = ln2 * log2sumexp2_n(betaT)
    float m = beta[0];
    #pragma unroll
    for (int r = 1; r < 8; ++r) m = fmaxf(m, beta[r]);
    red[w][half][bl] = m;
    __syncthreads();
    float gm = NEGINF;
    #pragma unroll
    for (int ww = 0; ww < 4; ++ww)
        #pragma unroll
        for (int h = 0; h < 2; ++h) gm = fmaxf(gm, red[ww][h][bl]);
    float s = 0.f;
    #pragma unroll
    for (int r = 0; r < 8; ++r) s += EXP2(beta[r] - gm);
    __syncthreads();
    red[w][half][bl] = s;
    __syncthreads();
    if (tid < 16) {
        float tot = 0.f;
        #pragma unroll
        for (int ww = 0; ww < 4; ++ww)
            #pragma unroll
            for (int h = 0; h < 2; ++h) tot += red[ww][h][tid];
        out[blockIdx.x * 16 + tid] = LN2 * (gm + LOG2(tot));
    }
}

// ---------------------------------------------------------------------------
// fac: aligned forward, also in the log2 domain (trans/emissions pre-scaled by
// log2e so the per-step chain has zero multiplies).  The global x row is
// prefetched one step ahead into a register before the barriers.
// One block per batch, one thread per target position l.
// out[b] -= ln2 * betaT[b, target_size[b]-1].
// ---------------------------------------------------------------------------
__global__ __launch_bounds__(256) void asg_fac(const float* __restrict__ input,
                                               const int* __restrict__ target,
                                               const int* __restrict__ tsize,
                                               const float* __restrict__ trans,
                                               float* __restrict__ out) {
    __shared__ float xrow[N_CLS];     // x[b,t,:] * log2e
    __shared__ float aprev[L_TGT];

    const int b = blockIdx.x;
    const int l = threadIdx.x;
    const int tl = target[b * L_TGT + l];
    const int tp = (l > 0) ? target[b * L_TGT + l - 1] : 0;
    const float tself = trans[tl * N_CLS + tl] * LOG2E;
    const float tmove = trans[tl * N_CLS + tp] * LOG2E;
    const float* xb = input + (size_t)b * T_LEN * N_CLS;

    float beta = (l == 0) ? xb[tl] * LOG2E : NEGINF;

    // prime the x-pipeline with row t = 1
    float xreg = (l < N_CLS) ? xb[(size_t)1 * N_CLS + l] * LOG2E : 0.f;

    for (int t = 1; t < T_LEN; ++t) {
        aprev[l] = beta;
        if (l < N_CLS) xrow[l] = xreg;
        // issue prefetch of row t+1 before the barrier (clamped on last iter)
        int tn = (t + 1 < T_LEN) ? (t + 1) : (T_LEN - 1);
        float xnext = (l < N_CLS) ? xb[(size_t)tn * N_CLS + l] * LOG2E : 0.f;
        __syncthreads();
        float emit = xrow[tl];
        float stay = beta + tself;
        float move = (l == 0) ? NEGINF : aprev[l - 1] + tmove;
        float mx = fmaxf(stay, move);
        float mn = fminf(stay, move);
        // log2addexp2: mx + log2(1 + exp2(mn-mx)); args safe for raw ops
        beta = emit + mx + LOG2(1.0f + EXP2(mn - mx));
        __syncthreads();
        xreg = xnext;
    }

    if (l == tsize[b] - 1) out[b] -= LN2 * beta;
}

extern "C" void kernel_launch(void* const* d_in, const int* in_sizes, int n_in,
                              void* d_out, int out_size, void* d_ws, size_t ws_size,
                              hipStream_t stream) {
    const float* input  = (const float*)d_in[0];
    const int*   target = (const int*)d_in[1];
    const int*   tsize  = (const int*)d_in[2];
    const float* trans  = (const float*)d_in[3];
    float* out = (float*)d_out;
    _Float16* eA = (_Float16*)d_ws;   // 4096 halfs = 8 KB

    asg_prep_eA<<<16, 256, 0, stream>>>(trans, eA);
    asg_fcc<<<8, 128, 0, stream>>>(input, eA, out);
    asg_fac<<<128, 256, 0, stream>>>(input, target, tsize, trans, out);
    (void)in_sizes; (void)n_in; (void)out_size; (void)ws_size;
}
